// HeteroGIN_45792941310086
// MI455X (gfx1250) — matrix operand order repaired and checked by reading, not compiled, not run on hardware
//
#include <hip/hip_runtime.h>

// ---------------------------------------------------------------------------
// HeteroGIN forward for MI455X (gfx1250, wave32, WMMA).
// Scatter stage: native global_atomic_add_f32. MLP stage: bf16 WMMA
// (v_wmma_f32_16x16x32_bf16) with f32 accumulation, fused bias/ReLU/LayerNorm.
// ---------------------------------------------------------------------------

#define DEVC __device__ __forceinline__

typedef __attribute__((ext_vector_type(16))) __bf16 bf16x16;
typedef __attribute__((ext_vector_type(8)))  __bf16 bf16x8;
typedef __attribute__((ext_vector_type(8)))  float  f32x8;

static constexpr int C   = 128;
static constexpr int C2  = 256;
static constexpr int NUc = 200000;
static constexpr int NIc = 200000;
static constexpr int Ec  = 500000;
static constexpr int Lc  = 2;

DEVC __bf16 f2bf(float f) {
  union { float f; unsigned u; } in; in.f = f;
  unsigned r = in.u + 0x7FFFu + ((in.u >> 16) & 1u);   // round-to-nearest-even
  union { unsigned short s; __bf16 b; } out;
  out.s = (unsigned short)(r >> 16);
  return out.b;
}

DEVC f32x8 wmma_bf16(bf16x16 a, bf16x16 b, f32x8 c) {
  // D = A(16x32 bf16) * B(32x16 bf16) + C(16x16 f32)
  return __builtin_amdgcn_wmma_f32_16x16x32_bf16(
      false, a, false, b, (short)0, c, false, false);
}

// A-matrix (16x32 bf16) fragment from LDS, per ISA layout:
// lane L: m = L&15, hi = L>>4; halves 0..7 -> K = k0+hi*8+{0..7},
//                              halves 8..15 -> K = k0+16+hi*8+{0..7}
DEVC bf16x16 frag_a_lds(const __bf16* base, int pitch, int mtile, int k0, int lane) {
  int m  = mtile * 16 + (lane & 15);
  int hi = lane >> 4;
  const __bf16* p = base + m * pitch + k0 + hi * 8;
  bf16x8 lo = *reinterpret_cast<const bf16x8*>(p);
  bf16x8 hh = *reinterpret_cast<const bf16x8*>(p + 16);
  bf16x16 r;
#pragma unroll
  for (int i = 0; i < 8; ++i) { r[i] = lo[i]; r[8 + i] = hh[i]; }
  return r;
}

// B-matrix (32x16 bf16) fragment from global, weights pre-transposed to [N][K]:
// lane L: n = L&15, hi = L>>4; halves 0..7 -> K = k0+hi*8+{0..7}, halves 8..15 -> K = k0+16+hi*8+{0..7}
DEVC bf16x16 frag_b_glb(const __bf16* wT, int K, int ntile, int k0, int lane) {
  int n  = ntile * 16 + (lane & 15);
  int hi = lane >> 4;
  const __bf16* p = wT + (size_t)n * K + k0 + hi * 8;
  bf16x8 lo = *reinterpret_cast<const bf16x8*>(p);
  bf16x8 hh = *reinterpret_cast<const bf16x8*>(p + 16);
  bf16x16 r;
#pragma unroll
  for (int i = 0; i < 8; ++i) { r[i] = lo[i]; r[8 + i] = hh[i]; }
  return r;
}

// Native L2 fp32 atomic add (non-returning -> STOREcnt; s_endpgm waits idle).
DEVC void atomic_add_f32g(float* p, float v) {
  asm volatile("global_atomic_add_f32 %0, %1, off" :: "v"(p), "v"(v) : "memory");
}

// ---------------------------------------------------------------------------
__global__ __launch_bounds__(256) void zero_kernel(float4* p, long n) {
  long i      = (long)blockIdx.x * blockDim.x + threadIdx.x;
  long stride = (long)gridDim.x * blockDim.x;
  float4 z; z.x = z.y = z.z = z.w = 0.f;
  for (; i < n; i += stride) p[i] = z;
}

// Convert per-layer weights to bf16, transposed to [N][K] for B-fragment loads.
__global__ __launch_bounds__(256) void prep_weights(
    const float* __restrict__ w1, const float* __restrict__ w2,
    __bf16* __restrict__ w1T, __bf16* __restrict__ w2T) {
  int tid = blockIdx.x * blockDim.x + threadIdx.x;
  const int per = Lc * C * C2;                 // 65536 elements each
  if (tid < per) {                             // w1: [L][C][C2] -> w1T [L][C2][C]
    int l = tid / (C * C2);
    int rem = tid - l * (C * C2);
    int k = rem / C2;
    int n = rem - k * C2;
    w1T[(size_t)l * C2 * C + (size_t)n * C + k] = f2bf(w1[tid]);
  } else if (tid < 2 * per) {                  // w2: [L][C2][C] -> w2T [L][C][C2]
    int t = tid - per;
    int l = t / (C2 * C);
    int rem = t - l * (C2 * C);
    int k = rem / C;
    int n = rem - k * C;
    w2T[(size_t)l * C * C2 + (size_t)n * C2 + k] = f2bf(w2[t]);
  }
}

// Edge scatter: 32 lanes per edge, 4 channels per lane.
template <int GINE>
__global__ __launch_bounds__(256) void scatter_kernel(
    const float* __restrict__ x, const float* __restrict__ eattr,
    const int* __restrict__ ei, float* __restrict__ agg, int nE) {
  int tid  = blockIdx.x * blockDim.x + threadIdx.x;
  int edge = tid >> 5;
  if (edge >= nE) return;
  int c   = (tid & 31) << 2;
  int src = ei[edge];
  int dst = ei[nE + edge];
  const float4 v = *reinterpret_cast<const float4*>(x + (size_t)src * C + c);
  float m0 = v.x, m1 = v.y, m2 = v.z, m3 = v.w;
  if (GINE) {
    const float4 a = *reinterpret_cast<const float4*>(eattr + (size_t)edge * C + c);
    m0 = fmaxf(m0 + a.x, 0.f); m1 = fmaxf(m1 + a.y, 0.f);
    m2 = fmaxf(m2 + a.z, 0.f); m3 = fmaxf(m3 + a.w, 0.f);
  }
  float* d = agg + (size_t)dst * C + c;
  atomic_add_f32g(d + 0, m0);
  atomic_add_f32g(d + 1, m1);
  atomic_add_f32g(d + 2, m2);
  atomic_add_f32g(d + 3, m3);
}

// Fused node update: h = (1+eps)*x + agg ; out = MLP(h) [ + MLP(h2) if DUAL ];
// LayerNorm; ReLU. 16 nodes per block, 8 waves, WMMA GEMMs.
template <int DUAL>
__global__ __launch_bounds__(256) void node_update(
    const float* __restrict__ x_in,
    const float* __restrict__ agg0,
    const float* __restrict__ agg1,
    const __bf16* __restrict__ w1T,   // [C2][C] bf16
    const float*  __restrict__ b1,    // [C2]
    const __bf16* __restrict__ w2T,   // [C][C2] bf16
    const float*  __restrict__ b2,    // [C]
    const float*  __restrict__ lnw,
    const float*  __restrict__ lnb,
    const float*  __restrict__ eps, int ei0, int ei1,
    float* __restrict__ x_out, int N) {
  constexpr int MT = DUAL ? 2 : 1;   // M-tiles in A (rows 16..31 = GIN branch)
  constexpr int AM = MT * 16;
  constexpr int P1 = C + 8;          // LDS pitches (bf16 halves / floats), 16B-aligned
  constexpr int P2 = C2 + 8;
  constexpr int PF = C + 4;

  __shared__ __bf16 hA[AM * P1];
  __shared__ __bf16 yA[AM * P2];
  __shared__ float  outS[AM * PF];
  __shared__ float  redS[16][17];
  __shared__ float  redQ[16][17];
  __shared__ float  stats[16][2];

  const int tid   = threadIdx.x;
  const int wave  = tid >> 5;
  const int lane  = tid & 31;
  const int node0 = blockIdx.x * 16;

  __builtin_prefetch(w1T, 0, 3);     // global_prefetch_b8 (weights L2-resident)
  __builtin_prefetch(w2T, 0, 3);

  const float e0 = 1.0f + eps[ei0];
  const float e1 = DUAL ? (1.0f + eps[ei1]) : 0.0f;

  // Phase 1: build bf16 A-matrix h in LDS.
  for (int idx = tid; idx < 16 * (C / 4); idx += 256) {
    int r  = idx >> 5;
    int c4 = (idx & 31) << 2;
    int node = node0 + r; if (node >= N) node = N - 1;  // clamp (rows unused)
    const float4 xv = *reinterpret_cast<const float4*>(x_in + (size_t)node * C + c4);
    const float4 a0 = *reinterpret_cast<const float4*>(agg0 + (size_t)node * C + c4);
    hA[r * P1 + c4 + 0] = f2bf(e0 * xv.x + a0.x);
    hA[r * P1 + c4 + 1] = f2bf(e0 * xv.y + a0.y);
    hA[r * P1 + c4 + 2] = f2bf(e0 * xv.z + a0.z);
    hA[r * P1 + c4 + 3] = f2bf(e0 * xv.w + a0.w);
    if (DUAL) {
      const float4 a1 = *reinterpret_cast<const float4*>(agg1 + (size_t)node * C + c4);
      hA[(16 + r) * P1 + c4 + 0] = f2bf(e1 * xv.x + a1.x);
      hA[(16 + r) * P1 + c4 + 1] = f2bf(e1 * xv.y + a1.y);
      hA[(16 + r) * P1 + c4 + 2] = f2bf(e1 * xv.z + a1.z);
      hA[(16 + r) * P1 + c4 + 3] = f2bf(e1 * xv.w + a1.w);
    }
  }
  __syncthreads();

  // Phase 2: y = relu(h @ w1 + b1)  -> bf16 LDS. MT*16 output tiles over 8 waves.
  for (int t = wave; t < MT * 16; t += 8) {
    int mt = t >> 4, nt = t & 15;
    f32x8 acc = {};
#pragma unroll
    for (int ks = 0; ks < 4; ++ks) {                       // K = 128
      bf16x16 af = frag_a_lds(hA, P1, mt, ks * 32, lane);
      bf16x16 bf = frag_b_glb(w1T, C, nt, ks * 32, lane);
      acc = wmma_bf16(af, bf, acc);
    }
    int col  = nt * 16 + (lane & 15);
    float bb = b1[col];
    int mrow = mt * 16 + (lane >> 4) * 8;
#pragma unroll
    for (int j = 0; j < 8; ++j) {
      float v = acc[j] + bb;
      yA[(mrow + j) * P2 + col] = f2bf(v > 0.f ? v : 0.f);
    }
  }
  __syncthreads();

  // Phase 3: out = y @ w2 + b2 -> f32 LDS. MT*8 tiles over 8 waves.
  for (int t = wave; t < MT * 8; t += 8) {
    int mt = t >> 3, nt = t & 7;
    f32x8 acc = {};
#pragma unroll
    for (int ks = 0; ks < 8; ++ks) {                       // K = 256
      bf16x16 af = frag_a_lds(yA, P2, mt, ks * 32, lane);
      bf16x16 bf = frag_b_glb(w2T, C2, nt, ks * 32, lane);
      acc = wmma_bf16(af, bf, acc);
    }
    int col  = nt * 16 + (lane & 15);
    float bb = b2[col];
    int mrow = mt * 16 + (lane >> 4) * 8;
#pragma unroll
    for (int j = 0; j < 8; ++j)
      outS[(mrow + j) * PF + col] = acc[j] + bb;
  }
  __syncthreads();

  // Phase 4: (sum DUAL branches) + LayerNorm + ReLU + store.
  {
    int row = tid >> 4, seg = tid & 15;
    float s = 0.f, q = 0.f;
#pragma unroll
    for (int c = seg * 8; c < seg * 8 + 8; ++c) {
      float v = outS[row * PF + c];
      if (DUAL) v += outS[(16 + row) * PF + c];
      s += v; q += v * v;
    }
    redS[row][seg] = s; redQ[row][seg] = q;
  }
  __syncthreads();
  if (tid < 16) {
    float s = 0.f, q = 0.f;
#pragma unroll
    for (int i = 0; i < 16; ++i) { s += redS[tid][i]; q += redQ[tid][i]; }
    float mu  = s * (1.f / 128.f);
    float var = q * (1.f / 128.f) - mu * mu;
    stats[tid][0] = mu;
    stats[tid][1] = rsqrtf(var + 1e-5f);
  }
  __syncthreads();
  for (int idx = tid; idx < 16 * C; idx += 256) {
    int row = idx >> 7, c = idx & (C - 1);
    int node = node0 + row;
    if (node < N) {
      float v = outS[row * PF + c];
      if (DUAL) v += outS[(16 + row) * PF + c];
      v = (v - stats[row][0]) * stats[row][1] * lnw[c] + lnb[c];
      x_out[(size_t)node * C + c] = v > 0.f ? v : 0.f;
    }
  }
}

// ---------------------------------------------------------------------------
extern "C" void kernel_launch(void* const* d_in, const int* in_sizes, int n_in,
                              void* d_out, int out_size, void* d_ws, size_t ws_size,
                              hipStream_t stream) {
  (void)in_sizes; (void)n_in; (void)out_size; (void)ws_size;

  const float* x_user = (const float*)d_in[0];
  const float* x_item = (const float*)d_in[1];
  const float* ea_ui  = (const float*)d_in[2];
  const float* ea_iu  = (const float*)d_in[3];
  const int*   ei_ui  = (const int*)d_in[4];
  const int*   ei_iu  = (const int*)d_in[5];
  const int*   ei_uu  = (const int*)d_in[6];
  const float* w1     = (const float*)d_in[7];
  const float* b1     = (const float*)d_in[8];
  const float* w2     = (const float*)d_in[9];
  const float* b2     = (const float*)d_in[10];
  const float* lnw_u  = (const float*)d_in[11];
  const float* lnb_u  = (const float*)d_in[12];
  const float* lnw_i  = (const float*)d_in[13];
  const float* lnb_i  = (const float*)d_in[14];
  const float* eps    = (const float*)d_in[15];

  // Workspace layout: 3 contiguous aggregation buffers, then bf16 weights.
  float*  aggI  = (float*)d_ws;                        // [NI][C]
  float*  aggUg = aggI  + (size_t)NIc * C;             // [NU][C]
  float*  aggUu = aggUg + (size_t)NUc * C;             // [NU][C]
  __bf16* w1T   = (__bf16*)(aggUu + (size_t)NUc * C);  // [L][C2][C]
  __bf16* w2T   = w1T + (size_t)Lc * C * C2;           // [L][C][C2]

  // State lives in d_out: [x_user ; x_item]. Layer-1 update is in-place safe.
  float* outU = (float*)d_out;
  float* outI = outU + (size_t)NUc * C;

  prep_weights<<<(2 * Lc * C * C2 + 255) / 256, 256, 0, stream>>>(w1, w2, w1T, w2T);

  const long zero4   = ((long)(NIc + 2 * NUc) * C) / 4;     // float4 count
  const int  sblocks = (Ec * 32 + 255) / 256;

  for (int l = 0; l < Lc; ++l) {
    zero_kernel<<<2048, 256, 0, stream>>>((float4*)aggI, zero4);

    const float* xu = l ? outU : x_user;
    const float* xi = l ? outI : x_item;

    scatter_kernel<1><<<sblocks, 256, 0, stream>>>(xu, ea_ui, ei_ui, aggI,  Ec);
    scatter_kernel<1><<<sblocks, 256, 0, stream>>>(xi, ea_iu, ei_iu, aggUg, Ec);
    scatter_kernel<0><<<sblocks, 256, 0, stream>>>(xu, nullptr, ei_uu, aggUu, Ec);

    node_update<0><<<(NIc + 15) / 16, 256, 0, stream>>>(
        xi, aggI, nullptr,
        w1T + (size_t)l * C * C2, b1 + l * C2,
        w2T + (size_t)l * C * C2, b2 + l * C,
        lnw_i + l * C, lnb_i + l * C, eps, l * 3 + 0, 0, outI, NIc);

    node_update<1><<<(NUc + 15) / 16, 256, 0, stream>>>(
        xu, aggUg, aggUu,
        w1T + (size_t)l * C * C2, b1 + l * C2,
        w2T + (size_t)l * C * C2, b2 + l * C,
        lnw_u + l * C, lnb_u + l * C, eps, l * 3 + 1, l * 3 + 2, outU, NUc);
  }
}
